// ILA_10986526343542
// MI455X (gfx1250) — matrix operand order repaired and checked by reading, not compiled, not run on hardware
//
#include <hip/hip_runtime.h>
#include <hip/hip_bf16.h>

// ---------------------------------------------------------------------------
// MI455X (gfx1250) linear-attention pipeline, bf16 WMMA + async-LDS staging.
//   x:[8,256,16384]  Wq/Wk:[256,256] Wv:[512,256] Wo:[256,512]
// All matmuls via v_wmma_f32_16x16x32_bf16 (wave32, 16x16 tiles, K=32).
// GEMM tiles staged with GLOBAL_LOAD_ASYNC_TO_LDS_B128 (ASYNCcnt) — activations
// are kept in transposed [n][c] layouts so B-tiles are contiguous-K copies.
// ---------------------------------------------------------------------------

typedef __bf16 v16bf __attribute__((ext_vector_type(16)));
typedef __bf16 v8bf  __attribute__((ext_vector_type(8)));
typedef float  v8f   __attribute__((ext_vector_type(8)));

union FragBF { v16bf v; v8bf h[2]; };

#define WMMA_BF16(A, B, C) \
  __builtin_amdgcn_wmma_f32_16x16x32_bf16(false, (A), false, (B), (short)0, (C), false, false)

static __device__ __forceinline__ __bf16 f2bf(float f) {
  union { float f; unsigned u; } a; a.f = f;
  unsigned u = a.u + 0x7FFFu + ((a.u >> 16) & 1u);   // RNE
  union { unsigned short s; __bf16 b; } o; o.s = (unsigned short)(u >> 16);
  return o.b;
}
static __device__ __forceinline__ float bf2f(__bf16 b) {
  union { __bf16 b; unsigned short s; } i; i.b = b;
  union { unsigned u; float f; } o; o.u = ((unsigned)i.s) << 16;
  return o.f;
}

// Async 16-byte global->LDS copy (per lane). LDS generic pointer low 32 bits
// are the in-allocation byte offset (ISA 10.2 aperture mapping).
static __device__ __forceinline__ void async_copy_b128(const void* g, void* l) {
  unsigned lds = (unsigned)(uintptr_t)l;
  unsigned long long ga = (unsigned long long)(uintptr_t)g;
  asm volatile("global_load_async_to_lds_b128 %0, %1, off" :: "v"(lds), "v"(ga) : "memory");
}
static __device__ __forceinline__ void wait_async() {
  asm volatile("s_wait_asynccnt 0x0" ::: "memory");
}

#define NTOK 16384      // h*w
#define NB   8          // batch
#define SCALE_QK 0.4204482076268573f   // 32^-0.25

// ---------------------------------------------------------------------------
// x [b][256][n] f32  ->  xT [b][n][256] bf16   (tiled 32x32 transpose)
// ---------------------------------------------------------------------------
__global__ __launch_bounds__(256)
void transpose_x_kernel(const float* __restrict__ x, __bf16* __restrict__ xT) {
  __shared__ __bf16 t[32][33];
  const int n0 = blockIdx.x * 32, c0 = blockIdx.y * 32, b = blockIdx.z;
  const int tn = threadIdx.x & 31, tr = threadIdx.x >> 5;
#pragma unroll
  for (int r = 0; r < 4; ++r) {
    int c = tr + r * 8;
    t[c][tn] = f2bf(x[((size_t)b * 256 + c0 + c) * NTOK + n0 + tn]);
  }
  __syncthreads();
#pragma unroll
  for (int r = 0; r < 4; ++r) {
    int n = tr + r * 8;
    xT[((size_t)b * NTOK + n0 + n) * 256 + c0 + tn] = t[tn][n];
  }
}

__global__ void cvt_w_kernel(const float* __restrict__ Wq, const float* __restrict__ Wk,
                             const float* __restrict__ Wv, const float* __restrict__ Wo,
                             __bf16* __restrict__ Wqkvb, __bf16* __restrict__ Wob) {
  int i = blockIdx.x * 256 + threadIdx.x;           // < 262144 + 131072
  if (i < 262144) {                                  // Wqkv: [1024][256]
    float v;
    if      (i <  65536) v = Wq[i];
    else if (i < 131072) v = Wk[i - 65536];
    else                 v = Wv[i - 131072];
    Wqkvb[i] = f2bf(v);
  } else {
    int j = i - 262144;                              // Wo: [256][512]
    if (j < 131072) Wob[j] = f2bf(Wo[j]);
  }
}

__global__ void zero_kernel(float* __restrict__ p, int n) {
  int i = blockIdx.x * 256 + threadIdx.x;
  if (i < n) p[i] = 0.f;
}

// ---------------------------------------------------------------------------
// GEMM 1: QKV projection.  C[1024, n] = Wqkv[1024,256] * x[256, n]  (per batch)
// 256 thr = 8 waves (4x2), tile 128(M) x 128(N), K-step 32, async-LDS staging.
// Epilogue: rows 0-255   -> QfT[n][256] = (c + bq) * scale   (f32, v8f store)
//           rows 256-511 -> expK[c][n]  = exp((c+bk)*scale)  (bf16)
//           rows 512+    -> Vb[c][n]    = c + bv             (bf16)
// ---------------------------------------------------------------------------
__global__ __launch_bounds__(256)
void gemm_qkv_kernel(const __bf16* __restrict__ Wqkv, const __bf16* __restrict__ xT,
                     const float* __restrict__ bq, const float* __restrict__ bk,
                     const float* __restrict__ bv,
                     float* __restrict__ QfT, __bf16* __restrict__ expK,
                     __bf16* __restrict__ Vb) {
  const int n0 = blockIdx.x * 128;
  const int m0 = blockIdx.y * 128;
  const int b  = blockIdx.z;
  const int tid = threadIdx.x, lane = tid & 31, wave = tid >> 5;
  const int wm = wave >> 1, wn = wave & 1;
  const int l = lane & 15;  const bool hi = lane >= 16;

  __shared__ __align__(16) __bf16 As[128 * 32];   // [m][k]
  __shared__ __align__(16) __bf16 Bs[128 * 32];   // [n][k]

  v8f acc[2][4];
#pragma unroll
  for (int i = 0; i < 2; ++i)
#pragma unroll
    for (int j = 0; j < 4; ++j) acc[i][j] = 0.f;

  for (int kk = 0; kk < 256; kk += 32) {
    // async stage: A (weights, row-major) and B (xT rows, contiguous K)
#pragma unroll
    for (int q = tid * 2; q < tid * 2 + 2; ++q) {
      int r = q >> 2, part = q & 3;
      async_copy_b128(Wqkv + (size_t)(m0 + r) * 256 + kk + part * 8,
                      As + r * 32 + part * 8);
      async_copy_b128(xT + ((size_t)(b * NTOK + n0 + r)) * 256 + kk + part * 8,
                      Bs + r * 32 + part * 8);
    }
    if (kk + 32 < 256) {
      __builtin_prefetch(xT + ((size_t)(b * NTOK + n0 + (tid >> 1))) * 256 + kk + 32, 0, 1);
      __builtin_prefetch(Wqkv + (size_t)(m0 + (tid >> 1)) * 256 + kk + 32, 0, 1);
    }
    wait_async();
    __syncthreads();

    FragBF a[2], bb[4];
#pragma unroll
    for (int i = 0; i < 2; ++i) {   // A: lanes0-15 K={0-7,16-23}; lanes16-31 K={8-15,24-31}
      const __bf16* base = As + (wm * 32 + i * 16 + l) * 32 + (hi ? 8 : 0);
      a[i].h[0] = *(const v8bf*)base;
      a[i].h[1] = *(const v8bf*)(base + 16);
    }
#pragma unroll
    for (int j = 0; j < 4; ++j) {   // B: lanes0-15 K={0-7,16-23}; lanes16-31 K={8-15,24-31}
      const __bf16* base = Bs + (wn * 64 + j * 16 + l) * 32 + (hi ? 8 : 0);
      bb[j].h[0] = *(const v8bf*)base;
      bb[j].h[1] = *(const v8bf*)(base + 16);
    }
#pragma unroll
    for (int i = 0; i < 2; ++i)
#pragma unroll
      for (int j = 0; j < 4; ++j)
        acc[i][j] = WMMA_BF16(a[i].v, bb[j].v, acc[i][j]);
    __syncthreads();
  }

#pragma unroll
  for (int i = 0; i < 2; ++i) {
    int rs = m0 + wm * 32 + i * 16 + (hi ? 8 : 0);   // 8 consecutive M rows
    if (rs < 256) {
      v8f bqv = *(const v8f*)(bq + rs);
#pragma unroll
      for (int j = 0; j < 4; ++j) {
        int gcol = n0 + wn * 64 + j * 16 + l;
        v8f r = (acc[i][j] + bqv) * SCALE_QK;
        *(v8f*)(QfT + ((size_t)(b * NTOK + gcol)) * 256 + rs) = r;
      }
    } else if (rs < 512) {
      int r0 = rs - 256;          // exp(k): |k| <~ 1 -> max-subtraction unnecessary
#pragma unroll
      for (int j = 0; j < 4; ++j) {
        int gcol = n0 + wn * 64 + j * 16 + l;
#pragma unroll
        for (int e = 0; e < 8; ++e)
          expK[((size_t)b * 256 + r0 + e) * NTOK + gcol] =
              f2bf(__expf((acc[i][j][e] + bk[r0 + e]) * SCALE_QK));
      }
    } else {
      int r0 = rs - 512;
#pragma unroll
      for (int j = 0; j < 4; ++j) {
        int gcol = n0 + wn * 64 + j * 16 + l;
#pragma unroll
        for (int e = 0; e < 8; ++e)
          Vb[((size_t)b * 512 + r0 + e) * NTOK + gcol] = f2bf(acc[i][j][e] + bv[r0 + e]);
      }
    }
  }
}

// ---------------------------------------------------------------------------
// Q softmax over key-dim. Contiguous per position: QfT[n][256] -> QsT[n][256].
// ---------------------------------------------------------------------------
__global__ __launch_bounds__(256)
void qsoftmax_kernel(const float* __restrict__ QfT, __bf16* __restrict__ QsT) {
  int flat = blockIdx.x * 256 + threadIdx.x;        // < 8*16384*8
  int h = flat & 7;
  int n = (flat >> 3) & (NTOK - 1);
  int b = flat >> 17;
  const float* src = QfT + ((size_t)(b * NTOK + n)) * 256 + h * 32;
  float v[32], m = -1e30f;
#pragma unroll
  for (int d = 0; d < 32; ++d) { v[d] = src[d]; m = fmaxf(m, v[d]); }
  float s = 0.f;
#pragma unroll
  for (int d = 0; d < 32; ++d) { v[d] = __expf(v[d] - m); s += v[d]; }
  float inv = 1.f / s;
  __bf16* dst = QsT + ((size_t)(b * NTOK + n)) * 256 + h * 32;
#pragma unroll
  for (int d = 0; d < 32; ++d) dst[d] = f2bf(v[d] * inv);
}

// ---------------------------------------------------------------------------
// Denominator: den[b][row] = sum_n expK[b][row][n].  grid(8 chunks, 256, 8)
// ---------------------------------------------------------------------------
__global__ __launch_bounds__(256)
void densum_kernel(const __bf16* __restrict__ expK, float* __restrict__ den) {
  int chunk = blockIdx.x, row = blockIdx.y, b = blockIdx.z;
  int tid = threadIdx.x;
  const __bf16* src = expK + ((size_t)b * 256 + row) * NTOK + chunk * 2048 + tid;
  float s = 0.f;
#pragma unroll
  for (int i = 0; i < 8; ++i) s += bf2f(src[i * 256]);
  __shared__ float red[256];
  red[tid] = s; __syncthreads();
  for (int off = 128; off; off >>= 1) {
    if (tid < off) red[tid] += red[tid + off];
    __syncthreads();
  }
  if (tid == 0) atomicAdd(&den[b * 256 + row], red[0]);
}

// ---------------------------------------------------------------------------
// ctx numerator: ctx[b][h][32][64] += expK[32, n] * Vb[64, n]^T over n-slice.
// Contraction dim n contiguous for both operands -> fragments straight from
// global as b128 pairs. grid(16 slices, 8, 8); LDS + f32-atomic reduction.
// ---------------------------------------------------------------------------
__global__ __launch_bounds__(256)
void ctx_gemm_kernel(const __bf16* __restrict__ expK, const __bf16* __restrict__ Vb,
                     float* __restrict__ ctx) {
  const int slice = blockIdx.x, h = blockIdx.y, b = blockIdx.z;
  const int tid = threadIdx.x, lane = tid & 31, wave = tid >> 5;
  const int l = lane & 15;  const bool hi = lane >= 16;

  __shared__ float ctxs[32 * 64];
  for (int i = tid; i < 32 * 64; i += 256) ctxs[i] = 0.f;
  __syncthreads();

  v8f acc[2][4];
#pragma unroll
  for (int i = 0; i < 2; ++i)
#pragma unroll
    for (int j = 0; j < 4; ++j) acc[i][j] = 0.f;

  const __bf16* Kbase = expK + ((size_t)b * 256 + h * 32) * NTOK;
  const __bf16* Vbase = Vb   + ((size_t)b * 512 + h * 64) * NTOK;

  int nstart = slice * 1024 + wave * 128;           // 4 K-iters of 32 per wave
  for (int c = 0; c < 4; ++c) {
    int nb = nstart + c * 32;
    FragBF a[2], bb[4];
#pragma unroll
    for (int i = 0; i < 2; ++i) {
      const __bf16* base = Kbase + (size_t)(i * 16 + l) * NTOK + nb + (hi ? 8 : 0);
      a[i].h[0] = *(const v8bf*)base;
      a[i].h[1] = *(const v8bf*)(base + 16);
    }
#pragma unroll
    for (int j = 0; j < 4; ++j) {
      const __bf16* base = Vbase + (size_t)(j * 16 + l) * NTOK + nb + (hi ? 8 : 0);
      bb[j].h[0] = *(const v8bf*)base;
      bb[j].h[1] = *(const v8bf*)(base + 16);
    }
#pragma unroll
    for (int i = 0; i < 2; ++i)
#pragma unroll
      for (int j = 0; j < 4; ++j)
        acc[i][j] = WMMA_BF16(a[i].v, bb[j].v, acc[i][j]);
  }

#pragma unroll
  for (int i = 0; i < 2; ++i)
#pragma unroll
    for (int e = 0; e < 8; ++e) {
      int row = i * 16 + e + (hi ? 8 : 0);
#pragma unroll
      for (int j = 0; j < 4; ++j)
        atomicAdd(&ctxs[row * 64 + j * 16 + l], acc[i][j][e]);   // ds_add_f32
    }
  __syncthreads();
  float* g = ctx + ((size_t)b * 8 + h) * 32 * 64;
  for (int i = tid; i < 2048; i += 256) atomicAdd(&g[i], ctxs[i]);
}

// ---------------------------------------------------------------------------
// ctx finalize: divide by denominator + transpose -> ctxT[b][h][64][32] bf16
// ---------------------------------------------------------------------------
__global__ void ctxfin_kernel(const float* __restrict__ ctx, const float* __restrict__ den,
                              __bf16* __restrict__ ctxT) {
  int i = blockIdx.x * 256 + threadIdx.x;           // < 8*8*64*32
  int d = i & 31, e = (i >> 5) & 63, h = (i >> 11) & 7, b = i >> 14;
  float v = ctx[(((size_t)b * 8 + h) * 32 + d) * 64 + e] / den[b * 256 + h * 32 + d];
  ctxT[i] = f2bf(v);
}

// ---------------------------------------------------------------------------
// out64T[b][n][h*64+e] = sum_d ctxT[e][d] * Qs[d][n]   (M=64, K=32, one step)
// grid(64 n-tiles of 256, 8 heads, 8); B tile async-staged from QsT[n][256].
// Epilogue: 8 consecutive M rows per lane -> single v8bf (b128) store.
// ---------------------------------------------------------------------------
__global__ __launch_bounds__(256)
void attn_out_kernel(const __bf16* __restrict__ ctxT, const __bf16* __restrict__ QsT,
                     __bf16* __restrict__ out64T) {
  const int n0 = blockIdx.x * 256, h = blockIdx.y, b = blockIdx.z;
  const int tid = threadIdx.x, lane = tid & 31, wave = tid >> 5;
  const int l = lane & 15;  const bool hi = lane >= 16;

  __shared__ __align__(16) __bf16 Bs[256 * 32];     // [n][d]
#pragma unroll
  for (int q = tid * 4; q < tid * 4 + 4; ++q) {
    int n = q >> 2, part = q & 3;
    async_copy_b128(QsT + ((size_t)(b * NTOK + n0 + n)) * 256 + h * 32 + part * 8,
                    Bs + n * 32 + part * 8);
  }
  wait_async();
  __syncthreads();

  const __bf16* Abase = ctxT + ((size_t)b * 8 + h) * 64 * 32;
  FragBF a[4], bb[2];
#pragma unroll
  for (int i = 0; i < 4; ++i) {
    const __bf16* base = Abase + (i * 16 + l) * 32 + (hi ? 8 : 0);
    a[i].h[0] = *(const v8bf*)base;
    a[i].h[1] = *(const v8bf*)(base + 16);
  }
#pragma unroll
  for (int j = 0; j < 2; ++j) {
    const __bf16* base = Bs + (wave * 32 + j * 16 + l) * 32 + (hi ? 8 : 0);
    bb[j].h[0] = *(const v8bf*)base;
    bb[j].h[1] = *(const v8bf*)(base + 16);
  }
  v8f acc[4][2];
#pragma unroll
  for (int i = 0; i < 4; ++i)
#pragma unroll
    for (int j = 0; j < 2; ++j) {
      acc[i][j] = 0.f;
      acc[i][j] = WMMA_BF16(a[i].v, bb[j].v, acc[i][j]);
    }

#pragma unroll
  for (int i = 0; i < 4; ++i) {
    int rs = i * 16 + (hi ? 8 : 0);
#pragma unroll
    for (int j = 0; j < 2; ++j) {
      int col = n0 + wave * 32 + j * 16 + l;
      v8bf pk;
#pragma unroll
      for (int e = 0; e < 8; ++e) pk[e] = f2bf(acc[i][j][e]);
      *(v8bf*)(out64T + ((size_t)(b * NTOK + col)) * 512 + h * 64 + rs) = pk;
    }
  }
}

// ---------------------------------------------------------------------------
// GEMM 3: O-projection. y[256, n] = Wo[256,512] * out64[512, n] + bo.
// Same schema as gemm_qkv; K = 512 (16 K-steps); async-LDS staging from out64T.
// ---------------------------------------------------------------------------
__global__ __launch_bounds__(256)
void oproj_kernel(const __bf16* __restrict__ Wob, const __bf16* __restrict__ out64T,
                  const float* __restrict__ bo, float* __restrict__ y) {
  const int n0 = blockIdx.x * 128;
  const int m0 = blockIdx.y * 128;
  const int b  = blockIdx.z;
  const int tid = threadIdx.x, lane = tid & 31, wave = tid >> 5;
  const int wm = wave >> 1, wn = wave & 1;
  const int l = lane & 15;  const bool hi = lane >= 16;

  __shared__ __align__(16) __bf16 As[128 * 32];
  __shared__ __align__(16) __bf16 Bs[128 * 32];

  v8f acc[2][4];
#pragma unroll
  for (int i = 0; i < 2; ++i)
#pragma unroll
    for (int j = 0; j < 4; ++j) acc[i][j] = 0.f;

  for (int kk = 0; kk < 512; kk += 32) {
#pragma unroll
    for (int q = tid * 2; q < tid * 2 + 2; ++q) {
      int r = q >> 2, part = q & 3;
      async_copy_b128(Wob + (size_t)(m0 + r) * 512 + kk + part * 8,
                      As + r * 32 + part * 8);
      async_copy_b128(out64T + ((size_t)(b * NTOK + n0 + r)) * 512 + kk + part * 8,
                      Bs + r * 32 + part * 8);
    }
    if (kk + 32 < 512)
      __builtin_prefetch(out64T + ((size_t)(b * NTOK + n0 + (tid >> 1))) * 512 + kk + 32, 0, 1);
    wait_async();
    __syncthreads();

    FragBF a[2], bb[4];
#pragma unroll
    for (int i = 0; i < 2; ++i) {
      const __bf16* base = As + (wm * 32 + i * 16 + l) * 32 + (hi ? 8 : 0);
      a[i].h[0] = *(const v8bf*)base;
      a[i].h[1] = *(const v8bf*)(base + 16);
    }
#pragma unroll
    for (int j = 0; j < 4; ++j) {
      const __bf16* base = Bs + (wn * 64 + j * 16 + l) * 32 + (hi ? 8 : 0);
      bb[j].h[0] = *(const v8bf*)base;
      bb[j].h[1] = *(const v8bf*)(base + 16);
    }
#pragma unroll
    for (int i = 0; i < 2; ++i)
#pragma unroll
      for (int j = 0; j < 4; ++j)
        acc[i][j] = WMMA_BF16(a[i].v, bb[j].v, acc[i][j]);
    __syncthreads();
  }

#pragma unroll
  for (int i = 0; i < 2; ++i)
#pragma unroll
    for (int e = 0; e < 8; ++e) {
      int grow = m0 + wm * 32 + i * 16 + e + (hi ? 8 : 0);
#pragma unroll
      for (int j = 0; j < 4; ++j) {
        int gcol = n0 + wn * 64 + j * 16 + l;
        y[((size_t)b * 256 + grow) * NTOK + gcol] = acc[i][j][e] + bo[grow];
      }
    }
}

// ---------------------------------------------------------------------------
extern "C" void kernel_launch(void* const* d_in, const int* in_sizes, int n_in,
                              void* d_out, int out_size, void* d_ws, size_t ws_size,
                              hipStream_t stream) {
  (void)in_sizes; (void)n_in; (void)out_size; (void)ws_size;
  const float* x  = (const float*)d_in[0];
  const float* Wq = (const float*)d_in[1];
  const float* bq = (const float*)d_in[2];
  const float* Wk = (const float*)d_in[3];
  const float* bk = (const float*)d_in[4];
  const float* Wv = (const float*)d_in[5];
  const float* bv = (const float*)d_in[6];
  const float* Wo = (const float*)d_in[7];
  const float* bo = (const float*)d_in[8];
  float* y = (float*)d_out;

  char* p = (char*)d_ws;
  auto carve = [&](size_t bytes) -> char* {
    char* r = p; p += (bytes + 255) & ~(size_t)255; return r;
  };
  const size_t CN = (size_t)NB * 256 * NTOK;        // 33.5M elems
  __bf16* xT     = (__bf16*)carve(CN * 2);          // [b][n][256]
  __bf16* Wqkvb  = (__bf16*)carve(1024 * 256 * 2);
  __bf16* Wob    = (__bf16*)carve(256 * 512 * 2);
  float*  QfT    = (float*) carve(CN * 4);          // [b][n][256]
  __bf16* expK   = (__bf16*)carve(CN * 2);          // [b][256][n]
  __bf16* Vb     = (__bf16*)carve(CN * 2 * 2);      // [b][512][n]
  __bf16* QsT    = (__bf16*)carve(CN * 2);          // [b][n][256]
  __bf16* out64T = (__bf16*)carve(CN * 2 * 2);      // [b][n][512]
  float*  den    = (float*) carve((size_t)NB * 256 * 4);
  float*  ctx    = (float*) carve((size_t)NB * 8 * 32 * 64 * 4);
  __bf16* ctxT   = (__bf16*)carve((size_t)NB * 8 * 64 * 32 * 2);

  transpose_x_kernel<<<dim3(NTOK / 32, 8, NB), 256, 0, stream>>>(x, xT);
  cvt_w_kernel<<<(262144 + 131072) / 256, 256, 0, stream>>>(Wq, Wk, Wv, Wo, Wqkvb, Wob);
  zero_kernel<<<(NB * 256 + 255) / 256, 256, 0, stream>>>(den, NB * 256);
  zero_kernel<<<(NB * 8 * 32 * 64 + 255) / 256, 256, 0, stream>>>(ctx, NB * 8 * 32 * 64);

  gemm_qkv_kernel<<<dim3(NTOK / 128, 8, NB), 256, 0, stream>>>(
      Wqkvb, xT, bq, bk, bv, QfT, expK, Vb);
  qsoftmax_kernel<<<(NB * 8 * NTOK) / 256, 256, 0, stream>>>(QfT, QsT);
  densum_kernel<<<dim3(8, 256, NB), 256, 0, stream>>>(expK, den);
  ctx_gemm_kernel<<<dim3(16, 8, NB), 256, 0, stream>>>(expK, Vb, ctx);
  ctxfin_kernel<<<(NB * 8 * 64 * 32) / 256, 256, 0, stream>>>(ctx, den, ctxT);
  attn_out_kernel<<<dim3(NTOK / 256, 8, NB), 256, 0, stream>>>(ctxT, QsT, out64T);
  oproj_kernel<<<dim3(NTOK / 128, 2, NB), 256, 0, stream>>>(Wob, out64T, bo, y);
}
